// BNNConv2d_17377437680282
// MI455X (gfx1250) — 8-bit matrix operand order repaired and checked by reading, compiled, not run on hardware
//
#include <hip/hip_runtime.h>
#include <cstdint>


typedef int v8i __attribute__((ext_vector_type(8)));

#define BATCH   16
#define CIN     256
#define COUT    256
#define HH      64
#define WW      64
#define NELEM   (BATCH * CIN * HH * WW)
#define KTOT    (CIN * 9)
#define NCHUNK  (KTOT / 64)
#define PACKED_BYTES ((COUT / 16) * NCHUNK * 1024)
#define WREGION (4 * NCHUNK * 1024)
#define LDS_TOTAL (2 * WREGION)

#define WS_SUM   0
#define WS_W1P   256
#define WS_W2P   (256 + PACKED_BYTES)
#define WS_XQ    (256 + 2 * PACKED_BYTES)

__device__ __forceinline__ float alpha_from_sum(float s) {
    return 2.0f * s / ((float)NELEM * 11.269427669584644f);
}

__global__ __launch_bounds__(256) void abssum_kernel(const float* __restrict__ x,
                                                     float* __restrict__ sum) {
    __shared__ float sm[256];
    float acc = 0.0f;
    for (int i = blockIdx.x * blockDim.x + threadIdx.x; i < NELEM;
         i += gridDim.x * blockDim.x)
        acc += fabsf(x[i]);
    sm[threadIdx.x] = acc;
    __syncthreads();
    for (int s = 128; s > 0; s >>= 1) {
        if (threadIdx.x < (unsigned)s) sm[threadIdx.x] += sm[threadIdx.x + s];
        __syncthreads();
    }
    if (threadIdx.x == 0) atomicAdd(sum, sm[0]);
}

__global__ __launch_bounds__(256) void quantize_kernel(const float* __restrict__ x,
                                                       const float* __restrict__ sum,
                                                       int8_t* __restrict__ xq) {
    int idx = blockIdx.x * blockDim.x + threadIdx.x;
    if (idx >= NELEM) return;
    float alpha = alpha_from_sum(*sum);
    float v = x[idx] / alpha;
    v = fminf(fmaxf(v, -128.0f), 127.0f);
    int q = (int)rintf(v);
    int w = idx & 63;
    int h = (idx >> 6) & 63;
    int c = (idx >> 12) & 255;
    int b = idx >> 20;
    xq[(((b * HH + h) * WW + w) << 8) + c] = (int8_t)q;
}

__global__ __launch_bounds__(256) void pack_weights_kernel(const float* __restrict__ wt,
                                                           const float* __restrict__ s1,
                                                           const float* __restrict__ s2,
                                                           int8_t* __restrict__ w1p,
                                                           int8_t* __restrict__ w2p) {
    int idx = blockIdx.x * blockDim.x + threadIdx.x;
    if (idx >= COUT * KTOT) return;
    int cout = idx / KTOT;
    int k    = idx - cout * KTOT;
    int kpos = k >> 8;
    int cin  = k & 255;
    float w  = wt[(cout * CIN + cin) * 9 + kpos];
    int   g1 = (w > 0.0f) - (w < 0.0f);
    float r  = w - (float)g1 * s1[cout];
    int   g2 = (r > 0.0f) - (r < 0.0f);
    int chunk = k >> 6;
    int kk    = k & 63;
    int hi    = kk >> 5;
    int g     = (kk >> 4) & 1;
    int rest  = kk & 15;
    int v     = hi * 4 + (rest >> 2);
    int j     = rest & 3;
    int lane  = g * 16 + (cout & 15);
    long off  = ((long)((cout >> 4) * NCHUNK + chunk) << 10) + lane * 32 + v * 4 + j;
    w1p[off] = (int8_t)g1;
    w2p[off] = (int8_t)g2;
}

__global__ __launch_bounds__(256) void conv_wmma_kernel(const int8_t* __restrict__ xq,
                                                        const int8_t* __restrict__ w1p,
                                                        const int8_t* __restrict__ w2p,
                                                        const float* __restrict__ sum,
                                                        const float* __restrict__ s1,
                                                        const float* __restrict__ s2,
                                                        const float* __restrict__ bias,
                                                        float* __restrict__ out) {
    extern __shared__ int8_t smem[];

    const int tid  = threadIdx.x;
    const int lane = tid & 31;
    const int wv   = tid >> 5;
    const int cg   = blockIdx.x & 3;
    const int pt   = (blockIdx.x >> 2) * 8 + wv;
    const int p0   = pt << 4;
    const int b    = p0 >> 12;
    const int h    = (p0 >> 6) & 63;
    const int w0   = p0 & 63;
    const int ln   = lane & 15;
    const int g    = lane >> 4;

    {
        const int8_t* src1 = w1p + (long)cg * WREGION;
        const int8_t* src2 = w2p + (long)cg * WREGION;
        for (int k = 0; k < WREGION / (256 * 16); ++k) {
            unsigned off = (unsigned)((k * 256 + tid) * 16);
            unsigned lds1 = off;
            unsigned lds2 = off + WREGION;
            asm volatile("global_load_async_to_lds_b128 %0, %1, %2"
                         :: "v"(lds1), "v"(off), "s"(src1) : "memory");
            asm volatile("global_load_async_to_lds_b128 %0, %1, %2"
                         :: "v"(lds2), "v"(off), "s"(src2) : "memory");
        }
        asm volatile("s_wait_asynccnt 0x0" ::: "memory");
        __syncthreads();
    }

    v8i acc1[4], acc2[4];
#pragma unroll
    for (int t = 0; t < 4; ++t) {
        acc1[t] = (v8i){0, 0, 0, 0, 0, 0, 0, 0};
        acc2[t] = (v8i){0, 0, 0, 0, 0, 0, 0, 0};
    }

    union AU { unsigned long long q[4]; v8i v; };

    for (int kpos = 0; kpos < 9; ++kpos) {
        const int kh = kpos / 3;
        const int kw = kpos - kh * 3;
        const int hh = h + kh - 1;
        const int wx = w0 + ln + kw - 1;
        const bool valid = (unsigned)hh < (unsigned)HH && (unsigned)wx < (unsigned)WW;
        const int8_t* row = xq + ((((long)(b * HH + hh) * WW + wx) << 8) + g * 16);
#pragma unroll
        for (int cc = 0; cc < 4; ++cc) {
            const int chunk = kpos * 4 + cc;
            AU a;
            if (valid) {
                const int8_t* ap = row + (cc << 6);
                a.q[0] = *(const unsigned long long*)(ap + 0);
                a.q[1] = *(const unsigned long long*)(ap + 8);
                a.q[2] = *(const unsigned long long*)(ap + 32);
                a.q[3] = *(const unsigned long long*)(ap + 40);
            } else {
                a.q[0] = a.q[1] = a.q[2] = a.q[3] = 0ull;
            }
            const v8i A = a.v;
#pragma unroll
            for (int t = 0; t < 4; ++t) {
                const int boff = ((t * NCHUNK + chunk) << 10) + lane * 32;
                const v8i B1 = *(const v8i*)(smem + boff);
                const v8i B2 = *(const v8i*)(smem + WREGION + boff);
                acc1[t] = __builtin_amdgcn_wmma_i32_16x16x64_iu8(true, A, true, B1,
                                                                 acc1[t], false, false);
                acc2[t] = __builtin_amdgcn_wmma_i32_16x16x64_iu8(true, A, true, B2,
                                                                 acc2[t], false, false);
            }
        }
    }

    const float alpha = alpha_from_sum(*sum);
#pragma unroll
    for (int t = 0; t < 4; ++t) {
        const int cout = (cg * 4 + t) * 16 + ln;
        const float sc1 = s1[cout] * alpha;
        const float sc2 = s2[cout] * alpha;
        const float bb  = bias[cout];
        const long obase = (((long)(b * COUT + cout) * HH + h) * WW) + w0;
#pragma unroll
        for (int r = 0; r < 8; ++r) {
            const int m = r + 8 * g;
            out[obase + m] = sc1 * (float)acc1[t][r] + sc2 * (float)acc2[t][r] + bb;
        }
    }
}

extern "C" void kernel_launch(void* const* d_in, const int* in_sizes, int n_in,
                              void* d_out, int out_size, void* d_ws, size_t ws_size,
                              hipStream_t stream) {
    const float* x    = (const float*)d_in[0];
    const float* wt   = (const float*)d_in[1];
    const float* bias = (const float*)d_in[2];
    const float* s1   = (const float*)d_in[3];
    const float* s2   = (const float*)d_in[4];
    float* out        = (float*)d_out;

    char* ws          = (char*)d_ws;
    float*  sum       = (float*)(ws + WS_SUM);
    int8_t* w1p       = (int8_t*)(ws + WS_W1P);
    int8_t* w2p       = (int8_t*)(ws + WS_W2P);
    int8_t* xq        = (int8_t*)(ws + WS_XQ);

    hipMemsetAsync(sum, 0, sizeof(float), stream);

    abssum_kernel<<<1024, 256, 0, stream>>>(x, sum);
    quantize_kernel<<<NELEM / 256, 256, 0, stream>>>(x, sum, xq);
    pack_weights_kernel<<<(COUT * KTOT + 255) / 256, 256, 0, stream>>>(wt, s1, s2,
                                                                       w1p, w2p);
    conv_wmma_kernel<<<2048, 256, LDS_TOTAL, stream>>>(xq, w1p, w2p, sum, s1, s2,
                                                       bias, out);
}
